// LinearQwen3MoeSparseMoeBlock_2181843386483
// MI455X (gfx1250) — matrix operand order repaired and checked by reading, not compiled, not run on hardware
//
#include <hip/hip_runtime.h>

// ---------------- problem constants ----------------
constexpr int kH = 2048;   // hidden
constexpr int kI = 768;    // moe intermediate
constexpr int kE = 64;     // experts
constexpr int kT = 1024;   // tokens (B*S)
constexpr int kTopK = 8;

// ---------------- CDNA5 WMMA types ----------------
typedef __attribute__((ext_vector_type(16))) __bf16   bf16x16;
typedef __attribute__((ext_vector_type(2)))  __bf16   bf16x2;
typedef __attribute__((ext_vector_type(8)))  float    f32x8;
typedef __attribute__((ext_vector_type(8)))  unsigned int u32x8;

// Native f32->bf16 (RNE); packed pair -> one dword. Should lower to
// v_cvt_pk_bf16_f32 instead of a 4-op integer RNE sequence.
static __device__ __forceinline__ unsigned int packbf2(float lo, float hi) {
  bf16x2 v = { (__bf16)lo, (__bf16)hi };
  return __builtin_bit_cast(unsigned int, v);
}
static __device__ __forceinline__ unsigned short f2bf_u16(float f) {
  __bf16 b = (__bf16)f;
  return __builtin_bit_cast(unsigned short, b);
}
static __device__ __forceinline__ f32x8 wmma_bf16(const u32x8& a, const u32x8& b, f32x8 c) {
  return __builtin_amdgcn_wmma_f32_16x16x32_bf16(
      false, __builtin_bit_cast(bf16x16, a),
      false, __builtin_bit_cast(bf16x16, b),
      (short)0, c, false, false);
}

// ---------------- workspace layout ----------------
// [cnt: kE ints][base: kE ints][idx: kE*kT ints][cw: kE*kT f32][hmid: kT*kTopK rows x kI bf16]
constexpr size_t OFF_CNT  = 0;
constexpr size_t OFF_BASE = OFF_CNT  + kE * sizeof(int);
constexpr size_t OFF_IDX  = OFF_BASE + kE * sizeof(int);
constexpr size_t OFF_CW   = OFF_IDX  + (size_t)kE * kT * sizeof(int);
constexpr size_t OFF_HMID = (OFF_CW + (size_t)kE * kT * sizeof(float) + 255) & ~(size_t)255;

// =====================================================================
// Router: logits -> top-8 (softmax denom cancels under renorm) -> lists
// =====================================================================
__global__ __launch_bounds__(64) void router_kernel(
    const float* __restrict__ x, const float* __restrict__ Wr,
    int* __restrict__ cnt, int* __restrict__ idx, float* __restrict__ cw)
{
  const int t = blockIdx.x;
  const int e = threadIdx.x;            // 0..63
  const float* xr = x + (size_t)t * kH;

  float acc = 0.f;
#pragma unroll 4
  for (int h = 0; h < kH; ++h) acc += xr[h] * Wr[(size_t)h * kE + e];

  __shared__ float pr[kE];
  pr[e] = acc;
  __syncthreads();

  if (e == 0) {
    float mx = -1e30f;
    for (int i = 0; i < kE; ++i) mx = fmaxf(mx, pr[i]);
    for (int i = 0; i < kE; ++i) pr[i] = __expf(pr[i] - mx);   // unnormalized probs
    int   sel[kTopK];
    float sw[kTopK];
    float wsum = 0.f;
    for (int k = 0; k < kTopK; ++k) {
      int   bj = 0; float bv = -1.f;
      for (int i = 0; i < kE; ++i) { float v = pr[i]; if (v > bv) { bv = v; bj = i; } }
      sel[k] = bj; sw[k] = bv; wsum += bv; pr[bj] = -1.f;
    }
    const float inv = 1.f / wsum;
    for (int k = 0; k < kTopK; ++k) {
      const int ee  = sel[k];
      const int pos = atomicAdd(&cnt[ee], 1);
      idx[(size_t)ee * kT + pos] = t;
      cw [(size_t)ee * kT + pos] = sw[k] * inv;
    }
  }
}

__global__ void scan_kernel(const int* __restrict__ cnt, int* __restrict__ basep) {
  if (threadIdx.x == 0) {
    int acc = 0;
    for (int e = 0; e < kE; ++e) { basep[e] = acc; acc += cnt[e]; }
  }
}

// =====================================================================
// Stage 1: per-expert  hmid = silu(x Wg) * (x Wu) * combw   (bf16 WMMA)
// grid (kE, kT/32), 512 threads = 16 waves; wave w -> i-tiles [3w,3w+3)
// =====================================================================
__global__ __launch_bounds__(512) void stage1_kernel(
    const float* __restrict__ x,  const float* __restrict__ Wg,
    const float* __restrict__ Wu, const int* __restrict__ cnt,
    const int* __restrict__ basep, const int* __restrict__ idx,
    const float* __restrict__ cw, unsigned short* __restrict__ hmid)
{
  const int e  = blockIdx.x;
  const int r0 = blockIdx.y * 32;
  const int ce = cnt[e];
  if (r0 >= ce) return;                       // uniform early exit
  const int be = basep[e];

  __shared__ unsigned int ldsA[32][17];       // 32 rows x 16 k-pairs (stride 17: no conflicts)
  __shared__ int   toks[32];
  __shared__ float cws[32];

  const int tid = threadIdx.x;
  if (tid < 32) {
    const int r = r0 + tid;
    toks[tid] = (r < ce) ? idx[(size_t)e * kT + r] : -1;
    cws [tid] = (r < ce) ? cw [(size_t)e * kT + r] : 0.f;
  }
  __syncthreads();

  const int wv   = tid >> 5;
  const int lane = tid & 31;
  const int n    = lane & 15;
  const int half = lane >> 4;

  // A-fill assignment for this thread
  const int fm = tid >> 4, fk = tid & 15;
  const int ftk = toks[fm];
  const float* fx = (ftk >= 0) ? (x + (size_t)ftk * kH) : nullptr;

  const float* wgE = Wg + (size_t)e * kH * kI;
  const float* wuE = Wu + (size_t)e * kH * kI;

  f32x8 cg[2][3], cu[2][3];
#pragma unroll
  for (int s = 0; s < 2; ++s)
#pragma unroll
    for (int it = 0; it < 3; ++it) {
      cg[s][it] = f32x8{0,0,0,0,0,0,0,0};
      cu[s][it] = f32x8{0,0,0,0,0,0,0,0};
    }

  for (int k0 = 0; k0 < kH; k0 += 32) {
    __syncthreads();
    { // stage x tile -> LDS as packed bf16 pairs
      unsigned int pv = 0;
      if (fx) {
        const float* p = fx + k0 + 2 * fk;
        pv = packbf2(p[0], p[1]);
      }
      ldsA[fm][fk] = pv;
    }
    __syncthreads();

    // A fragments: VGPR v packs K=half*16+2v,2v+1
    u32x8 a0, a1;
#pragma unroll
    for (int v = 0; v < 8; ++v) {
      const int p = half * 8 + v;
      a0[v] = ldsA[n][p];
      a1[v] = ldsA[16 + n][p];
    }

#pragma unroll
    for (int it = 0; it < 3; ++it) {
      const int icol0 = (wv * 3 + it) * 16;
      u32x8 bg, bu;
#pragma unroll
      for (int v = 0; v < 8; ++v) {      // B layout: lane=N, VGPR v packs K=half*16+2v,2v+1
        const size_t row = (size_t)(k0 + half * 16 + 2 * v);
        const float* pg = wgE + row * kI + icol0 + n;
        const float* pu = wuE + row * kI + icol0 + n;
        bg[v] = packbf2(pg[0], pg[kI]);
        bu[v] = packbf2(pu[0], pu[kI]);
      }
      cg[0][it] = wmma_bf16(a0, bg, cg[0][it]);
      cg[1][it] = wmma_bf16(a1, bg, cg[1][it]);
      cu[0][it] = wmma_bf16(a0, bu, cu[0][it]);
      cu[1][it] = wmma_bf16(a1, bu, cu[1][it]);
    }
  }

  // epilogue: silu(g)*u*combw -> bf16 hmid rows
#pragma unroll
  for (int s = 0; s < 2; ++s)
#pragma unroll
    for (int it = 0; it < 3; ++it) {
      const int icol0 = (wv * 3 + it) * 16;
#pragma unroll
      for (int j = 0; j < 8; ++j) {      // C layout: VGPR j -> M = j + 8*half (+16*s)
        const int m = 16 * s + 8 * half + j;
        const int r = r0 + m;
        if (r < ce) {
          const float g = cg[s][it][j], u = cu[s][it][j];
          const float hv = (g / (1.f + __expf(-g))) * u * cws[m];
          hmid[(size_t)(be + r) * kI + icol0 + n] = f2bf_u16(hv);
        }
      }
    }
}

// =====================================================================
// Stage 2: out[t,:] += hmid_row @ Wd_e ; scatter via f32 atomics
// grid (kE, kT/32), 512 threads = 16 waves; wave w -> n-tiles [8w,8w+8)
// =====================================================================
__global__ __launch_bounds__(512) void stage2_kernel(
    const unsigned short* __restrict__ hmid, const float* __restrict__ Wd,
    const int* __restrict__ cnt, const int* __restrict__ basep,
    const int* __restrict__ idx, float* __restrict__ out)
{
  const int e  = blockIdx.x;
  const int r0 = blockIdx.y * 32;
  const int ce = cnt[e];
  if (r0 >= ce) return;
  const int be = basep[e];

  __shared__ unsigned int ldsA[32][17];
  __shared__ int toks[32];

  const int tid = threadIdx.x;
  if (tid < 32) {
    const int r = r0 + tid;
    toks[tid] = (r < ce) ? idx[(size_t)e * kT + r] : -1;
  }
  __syncthreads();

  const int wv   = tid >> 5;
  const int lane = tid & 31;
  const int n    = lane & 15;
  const int half = lane >> 4;

  const int fm = tid >> 4, fk = tid & 15;
  const int fr = r0 + fm;
  const unsigned short* fh = (fr < ce) ? (hmid + (size_t)(be + fr) * kI) : nullptr;

  const float* wdE = Wd + (size_t)e * kI * kH;

  f32x8 c[2][8];
#pragma unroll
  for (int s = 0; s < 2; ++s)
#pragma unroll
    for (int it = 0; it < 8; ++it) c[s][it] = f32x8{0,0,0,0,0,0,0,0};

  for (int k0 = 0; k0 < kI; k0 += 32) {
    __syncthreads();
    {
      unsigned int pv = 0;
      if (fh) pv = *(const unsigned int*)(fh + k0 + 2 * fk);   // bf16 pair, 4B aligned
      ldsA[fm][fk] = pv;
    }
    __syncthreads();

    u32x8 a0, a1;
#pragma unroll
    for (int v = 0; v < 8; ++v) {
      const int p = half * 8 + v;
      a0[v] = ldsA[n][p];
      a1[v] = ldsA[16 + n][p];
    }

#pragma unroll
    for (int it = 0; it < 8; ++it) {
      const int n0 = (wv * 8 + it) * 16;
      u32x8 b;
#pragma unroll
      for (int v = 0; v < 8; ++v) {
        const size_t row = (size_t)(k0 + half * 16 + 2 * v);
        const float* pd = wdE + row * kH + n0 + n;
        b[v] = packbf2(pd[0], pd[kH]);
      }
      c[0][it] = wmma_bf16(a0, b, c[0][it]);
      c[1][it] = wmma_bf16(a1, b, c[1][it]);
    }
  }

#pragma unroll
  for (int s = 0; s < 2; ++s)
#pragma unroll
    for (int it = 0; it < 8; ++it) {
      const int n0 = (wv * 8 + it) * 16;
#pragma unroll
      for (int j = 0; j < 8; ++j) {
        const int m = 16 * s + 8 * half + j;
        const int r = r0 + m;
        if (r < ce) {
          const int t = toks[m];
          atomicAdd(&out[(size_t)t * kH + n0 + n], c[s][it][j]);
        }
      }
    }
}

// =====================================================================
extern "C" void kernel_launch(void* const* d_in, const int* in_sizes, int n_in,
                              void* d_out, int out_size, void* d_ws, size_t ws_size,
                              hipStream_t stream) {
  const float* x  = (const float*)d_in[0];   // [1,1024,2048]
  const float* Wr = (const float*)d_in[1];   // [H,E]
  const float* Wg = (const float*)d_in[2];   // [E,H,I]
  const float* Wu = (const float*)d_in[3];   // [E,H,I]
  const float* Wd = (const float*)d_in[4];   // [E,I,H]
  float* out = (float*)d_out;

  char* ws = (char*)d_ws;
  int*            cnt   = (int*)  (ws + OFF_CNT);
  int*            basep = (int*)  (ws + OFF_BASE);
  int*            idx   = (int*)  (ws + OFF_IDX);
  float*          cw    = (float*)(ws + OFF_CW);
  unsigned short* hmid  = (unsigned short*)(ws + OFF_HMID);

  hipMemsetAsync(cnt, 0, kE * sizeof(int), stream);
  hipMemsetAsync(out, 0, (size_t)kT * kH * sizeof(float), stream);

  router_kernel<<<kT, 64, 0, stream>>>(x, Wr, cnt, idx, cw);
  scan_kernel<<<1, 64, 0, stream>>>(cnt, basep);
  stage1_kernel<<<dim3(kE, kT / 32), 512, 0, stream>>>(x, Wg, Wu, cnt, basep, idx, cw, hmid);
  stage2_kernel<<<dim3(kE, kT / 32), 512, 0, stream>>>(hmid, Wd, cnt, basep, idx, out);
}
